// MyAttention_15857019257098
// MI455X (gfx1250) — matrix operand order repaired and checked by reading, not compiled, not run on hardware
//
#include <hip/hip_runtime.h>
#include <math.h>

// ---------------------------------------------------------------------------
// MI455X / gfx1250 implementation. wave32. Dense GEMMs: v_wmma_f32_16x16x32_bf16
// (bf16 in, f32 acc), 16x64 output strip per wave (4 back-to-back WMMAs per
// K-chunk). Global->LDS tile staging via the Tensor Data Mover
// (tensor_load_to_lds + s_wait_tensorcnt). Attention uses the argmax-mask
// algebra: out = (1/Z) * sum_{k: score==rowmax} vs[k], flash-style two passes,
// never materializing the 2048x2048 attn matrix.
// ---------------------------------------------------------------------------

typedef __attribute__((ext_vector_type(16))) __bf16 v16bf;
typedef __attribute__((ext_vector_type(8)))  float  v8f;
typedef __attribute__((ext_vector_type(4)))  unsigned int v4u;
typedef __attribute__((ext_vector_type(8)))  int v8i;
typedef __attribute__((ext_vector_type(4)))  int v4i;

static __device__ __forceinline__ v8f wmma_bf16(v16bf a, v16bf b, v8f c) {
  return __builtin_amdgcn_wmma_f32_16x16x32_bf16(
      /*neg_a=*/false, a, /*neg_b=*/false, b,
      /*c_mod=*/(short)0, c, /*reuse_a=*/false, /*reuse_b=*/false);
}

// Tensor Data Mover: DMA a 2D fp32 tile (rows x cols, row stride in elements)
// from global memory into LDS (contiguous row-major). D# per CDNA5 ISA
// 8.3/8.4: group0 = {count/type/lds_addr/global_addr}, group1 = dims/strides.
// clang-23 toolchain: 6-arg builtin (g0, g1, g2, g3, g_extra, cpol).
static __device__ __forceinline__ void tdm_load_2d(const float* gsrc,
                                                   float* ldst,
                                                   int cols, int rows,
                                                   int row_stride)
{
  unsigned long long ga = (unsigned long long)(uintptr_t)gsrc;
  unsigned int lo = (unsigned int)(uintptr_t)ldst;  // flat low 32 = LDS addr
  v4u g0;
  g0[0] = 1u;                                        // count=1 (valid user D#)
  g0[1] = lo;                                        // lds_addr
  g0[2] = (unsigned int)(ga & 0xFFFFFFFFu);          // global_addr[31:0]
  g0[3] = (unsigned int)((ga >> 32) & 0x01FFFFFFu)   // global_addr[56:32]
        | (2u << 30);                                // type = 2 ("image")
  const unsigned int td0 = 0x40000000u;              // huge tensor_dim0 (no OOB)
  v8i g1;
  g1[0] = (int)(2u << 16);                           // data_size = 4 bytes
  g1[1] = (int)((td0 & 0xFFFFu) << 16);              // tensor_dim0[15:0]
  g1[2] = (int)(((td0 >> 16) & 0xFFFFu)              // tensor_dim0[31:16]
        | ((td0 & 0xFFFFu) << 16));                  // tensor_dim1[15:0]
  g1[3] = (int)(((td0 >> 16) & 0xFFFFu)              // tensor_dim1[31:16]
        | (((unsigned)cols & 0xFFFFu) << 16));       // tile_dim0
  g1[4] = (int)((unsigned)rows & 0xFFFFu);           // tile_dim1 (tile_dim2=0)
  g1[5] = row_stride;                                // tensor_dim0_stride[31:0]
  g1[6] = 0;                                         // stride0 hi / stride1 lo
  g1[7] = 0;
  v4i z4 = {0, 0, 0, 0};
  v8i z8 = {0, 0, 0, 0, 0, 0, 0, 0};
  __builtin_amdgcn_tensor_load_to_lds(g0, g1, z4, z4, z8, 0);
}

static __device__ __forceinline__ void tdm_wait() {
  __builtin_amdgcn_s_wait_tensorcnt(0);
}

// ---------------------------------------------------------------------------
// GEMM: C[M,N] = act(A[M,K] @ W[K,N] + bias (+ Cin)). One wave per 16x64
// output strip: A fragment reused across 4 WMMAs per 32-wide K chunk.
// ---------------------------------------------------------------------------
__global__ __launch_bounds__(32)
void gemm16x64_bf16(const float* __restrict__ A, const float* __restrict__ W,
                    const float* __restrict__ bias, const float* __restrict__ Cin,
                    float* __restrict__ C, int M, int N, int K, int relu)
{
  const int tiles_n = N >> 6;
  const int tm = blockIdx.x / tiles_n;
  const int n0 = (blockIdx.x - tm * tiles_n) * 64;
  __shared__ float sA[512];    // 16 x 32
  __shared__ float sB[2048];   // 32 x 64
  const int lane = threadIdx.x;
  const int g  = lane >> 4;
  const int mr = lane & 15;
  v8f acc[4] = {};
  if (Cin) {
#pragma unroll
    for (int j = 0; j < 4; j++)
#pragma unroll
      for (int r = 0; r < 8; r++)
        acc[j][r] = Cin[(size_t)(tm * 16 + r + 8 * g) * N + n0 + j * 16 + mr];
  }
  for (int k0 = 0; k0 < K; k0 += 32) {
    tdm_load_2d(&A[(size_t)(tm * 16) * K + k0], sA, 32, 16, K);
    tdm_load_2d(&W[(size_t)k0 * N + n0], sB, 64, 32, N);
    tdm_wait();
    __syncthreads();
    v16bf af{};
#pragma unroll
    for (int e = 0; e < 16; e++) {
      int v = e >> 1, h = e & 1;
      // 16-bit A 16x32 layout: lanes 0-15: v0..3->K0..7, v4..7->K16..23; +8 for hi half
      int ka = (v < 4) ? (2 * v + h + 8 * g) : (16 + 2 * (v - 4) + h + 8 * g);
      af[e] = (__bf16)sA[mr * 32 + ka];
    }
#pragma unroll
    for (int j = 0; j < 4; j++) {
      v16bf bf{};
#pragma unroll
      for (int e = 0; e < 16; e++) {
        int v = e >> 1, h = e & 1;
        // 16-bit B 32x16 layout: v0..3->K 2v+h+8g, v4..7->K 16+2(v-4)+h+8g, N = lane%16
        int kb = (v < 4) ? (2 * v + h + 8 * g) : (16 + 2 * (v - 4) + h + 8 * g);
        bf[e] = (__bf16)sB[kb * 64 + j * 16 + mr];
      }
      acc[j] = wmma_bf16(af, bf, acc[j]);
    }
    __syncthreads();
  }
#pragma unroll
  for (int j = 0; j < 4; j++)
#pragma unroll
    for (int r = 0; r < 8; r++) {
      float v = acc[j][r];
      if (bias) v += bias[n0 + j * 16 + mr];
      if (relu) v = fmaxf(v, 0.f);
      C[(size_t)(tm * 16 + r + 8 * g) * N + n0 + j * 16 + mr] = v;
    }
}

// ---------------------------------------------------------------------------
// Argmax-masked attention. One wave per (batch, 16 query rows).
// out[q,:] = (1/Z_q) * sum_{k: s_qk == max_k s_qk} vs[k,:],  s = (qs@ks^T)/16
// ---------------------------------------------------------------------------
__global__ __launch_bounds__(32)
void attn_argmax(const float* __restrict__ qs, const float* __restrict__ ks,
                 const float* __restrict__ vs, float* __restrict__ out, int L)
{
  const int b = blockIdx.y;
  const int qt = blockIdx.x;
  const int lane = threadIdx.x;
  const int g = lane >> 4, mr = lane & 15;
  __shared__ float sT[2048];    // 16 x 128 staging tile
  __shared__ float sAcc[2048];  // 16 rows x 128 V accumulator
  __shared__ float sZ[16];
  const float* Q  = qs + ((size_t)b * L + qt * 16) * 128;
  const float* Kb = ks + (size_t)b * L * 128;
  const float* Vb = vs + (size_t)b * L * 128;

  // Stage Q tile (TDM), build 4 A-fragments, keep in registers.
  tdm_load_2d(Q, sT, 2048, 1, 0);
  tdm_wait();
  __syncthreads();
  v16bf aq[4];
#pragma unroll
  for (int c4 = 0; c4 < 4; c4++) {
#pragma unroll
    for (int e = 0; e < 16; e++) {
      int v = e >> 1, h = e & 1;
      int ka = (v < 4) ? (2 * v + h + 8 * g) : (16 + 2 * (v - 4) + h + 8 * g);
      aq[c4][e] = (__bf16)sT[mr * 128 + c4 * 32 + ka];
    }
  }
  __syncthreads();

  float lm[8], lz[8];
#pragma unroll
  for (int r = 0; r < 8; r++) { lm[r] = -3.0e38f; lz[r] = 0.f; }
  const float sc_inv = 1.0f / 16.0f;   // 1/sqrt(E)
  const int NT = L >> 4;

  // ---- pass 1: per-row max + sum(exp) ----
  for (int kt = 0; kt < NT; kt++) {
    tdm_load_2d(Kb + (size_t)kt * 2048, sT, 2048, 1, 0);
    tdm_wait();
    __syncthreads();
    v8f sc = {};
#pragma unroll
    for (int c4 = 0; c4 < 4; c4++) {
      v16bf bk{};
#pragma unroll
      for (int e = 0; e < 16; e++) {
        int v = e >> 1, h = e & 1;
        // B[K=d][N=key] = ks[key][d]: key = kt*16+mr, d = c4*32 + (A-fragment K order)
        int kb = (v < 4) ? (2 * v + h + 8 * g) : (16 + 2 * (v - 4) + h + 8 * g);
        bk[e] = (__bf16)sT[mr * 128 + c4 * 32 + kb];
      }
      sc = wmma_bf16(aq[c4], bk, sc);
    }
#pragma unroll
    for (int r = 0; r < 8; r++) {
      float s = sc[r] * sc_inv;
      if (s <= lm[r]) { lz[r] += __expf(s - lm[r]); }
      else           { lz[r] = lz[r] * __expf(lm[r] - s) + 1.0f; lm[r] = s; }
    }
    __syncthreads();
  }
  // combine max/Z across the 16 lanes of each half (rows r+8g)
#pragma unroll
  for (int off = 1; off < 16; off <<= 1) {
#pragma unroll
    for (int r = 0; r < 8; r++) {
      float om = __shfl_xor(lm[r], off, 16);
      float oz = __shfl_xor(lz[r], off, 16);
      float nm = fmaxf(lm[r], om);
      lz[r] = lz[r] * __expf(lm[r] - nm) + oz * __expf(om - nm);
      lm[r] = nm;
    }
  }
  if (mr == 0) {
#pragma unroll
    for (int r = 0; r < 8; r++) sZ[r + 8 * g] = lz[r];
  }
  for (int i = 0; i < 64; i++) sAcc[i * 32 + lane] = 0.f;
  __syncthreads();

  // ---- pass 2: accumulate vs rows where score hits the row max ----
  for (int kt = 0; kt < NT; kt++) {
    tdm_load_2d(Kb + (size_t)kt * 2048, sT, 2048, 1, 0);
    tdm_wait();
    __syncthreads();
    v8f sc = {};
#pragma unroll
    for (int c4 = 0; c4 < 4; c4++) {
      v16bf bk{};
#pragma unroll
      for (int e = 0; e < 16; e++) {
        int v = e >> 1, h = e & 1;
        int kb = (v < 4) ? (2 * v + h + 8 * g) : (16 + 2 * (v - 4) + h + 8 * g);
        bk[e] = (__bf16)sT[mr * 128 + c4 * 32 + kb];
      }
      sc = wmma_bf16(aq[c4], bk, sc);
    }
#pragma unroll
    for (int r = 0; r < 8; r++) {
      float s = sc[r] * sc_inv;
      if (s == lm[r]) {
        const float* vrow = Vb + (size_t)(kt * 16 + mr) * 128;
        for (int d = 0; d < 128; d++)
          atomicAdd(&sAcc[(r + 8 * g) * 128 + d], vrow[d]);
      }
    }
    __syncthreads();
  }
  for (int i = 0; i < 64; i++) {
    int idx = i * 32 + lane;
    int row = idx >> 7;
    out[((size_t)b * L + qt * 16 + row) * 128 + (idx & 127)] =
        sAcc[idx] / sZ[row];
  }
}

// ---------------------------------------------------------------------------
// Small helper kernels
// ---------------------------------------------------------------------------
__global__ __launch_bounds__(256)
void embed_k(const int* __restrict__ x, const float* __restrict__ emb,
             float* __restrict__ a_x)
{
  int row = blockIdx.x, e = threadIdx.x;
  a_x[(size_t)row * 256 + e] = emb[(size_t)x[row] * 256 + e];
}

__global__ __launch_bounds__(256)
void master_k(const float* __restrict__ master, const float* __restrict__ W,
              const float* __restrict__ bias, float* __restrict__ m)
{
  int b = blockIdx.x, e = threadIdx.x;
  float acc = bias[e];
  for (int i = 0; i < 64; i++) acc += master[b * 64 + i] * W[i * 256 + e];
  m[b * 256 + e] = acc;
}

__global__ __launch_bounds__(256)
void timegather_k(const int* __restrict__ timep, const float* __restrict__ table,
                  float* __restrict__ raw)
{
  int row = blockIdx.x, e = threadIdx.x;
  int t = -timep[row];
  raw[(size_t)row * 256 + e] = table[(size_t)t * 256 + e];
}

__global__ __launch_bounds__(256)
void gate_k(float* __restrict__ a_x, const float* __restrict__ kg,
            const float* __restrict__ m, long total)
{
  long i = (long)blockIdx.x * 256 + threadIdx.x;
  if (i >= total) return;
  int b = (int)(i / (2048L * 256));
  int e = (int)(i & 255);
  float mv = m[b * 256 + e];
  float sig = 1.f / (1.f + __expf(-(kg[i] * mv)));
  a_x[i] += sig * mv;
}

__global__ __launch_bounds__(256)
void addet_k(float* __restrict__ dst, const float* __restrict__ e_t, long total)
{
  long i = (long)blockIdx.x * 256 + threadIdx.x;
  if (i >= total) return;
  long row = i >> 8;
  int e = (int)(i & 255);
  int b = (int)(row >> 11);
  int l = (int)(row & 2047);
  int t = l >> 4;   // broadcast over C=16
  dst[i] += e_t[((size_t)b * 128 + t) * 256 + e];
}

__global__ __launch_bounds__(256)
void pawl2_k(const float* __restrict__ h, const float* __restrict__ w,
             const float* __restrict__ bias, float* __restrict__ logits, int Mr)
{
  int row = blockIdx.x * 256 + threadIdx.x;
  if (row >= Mr) return;
  float acc = bias[0];
  for (int e = 0; e < 256; e++) acc += h[(size_t)row * 256 + e] * w[e];
  logits[row] = acc;
}

__global__ __launch_bounds__(256)
void pool_final_k(const float* __restrict__ logits, const float* __restrict__ kx,
                  const float* __restrict__ w_out, float* __restrict__ out, int L)
{
  int b = blockIdx.x, t = threadIdx.x;
  __shared__ float red[256];
  float m = -3.0e38f;
  for (int l = t; l < L; l += 256) m = fmaxf(m, logits[b * L + l]);
  red[t] = m; __syncthreads();
  for (int s = 128; s > 0; s >>= 1) { if (t < s) red[t] = fmaxf(red[t], red[t + s]); __syncthreads(); }
  m = red[0]; __syncthreads();
  float z = 0.f;
  for (int l = t; l < L; l += 256) z += __expf(logits[b * L + l] - m);
  red[t] = z; __syncthreads();
  for (int s = 128; s > 0; s >>= 1) { if (t < s) red[t] += red[t + s]; __syncthreads(); }
  z = red[0]; __syncthreads();
  float acc = 0.f;  // pooled[b][e=t]
  for (int l = 0; l < L; l++)
    acc += __expf(logits[b * L + l] - m) * kx[((size_t)b * L + l) * 256 + t];
  acc /= z;
  red[t] = acc * w_out[t]; __syncthreads();
  for (int s = 128; s > 0; s >>= 1) { if (t < s) red[t] += red[t + s]; __syncthreads(); }
  if (t == 0) out[b] = red[0];
}

// ---------------------------------------------------------------------------
// Host-side orchestration
// ---------------------------------------------------------------------------
extern "C" void kernel_launch(void* const* d_in, const int* in_sizes, int n_in,
                              void* d_out, int out_size, void* d_ws, size_t ws_size,
                              hipStream_t stream)
{
  (void)in_sizes; (void)n_in; (void)out_size; (void)ws_size;
  constexpr int L = 2048, Mr = 16 * 2048;   // B*T*C rows

  // inputs in setup_inputs() dict order, depth-first
  const int*   x       = (const int*)  d_in[0];
  const float* master  = (const float*)d_in[1];
  const int*   timep   = (const int*)  d_in[2];
  const float* v_emb   = (const float*)d_in[3];
  const float* me_w    = (const float*)d_in[4];
  const float* me_b    = (const float*)d_in[5];
  const float* am_w1   = (const float*)d_in[6];
  const float* am_b1   = (const float*)d_in[7];
  const float* am_w2   = (const float*)d_in[8];
  const float* am_b2   = (const float*)d_in[9];
  const float* qm_w1   = (const float*)d_in[10];
  const float* qm_b1   = (const float*)d_in[11];
  const float* qm_w2   = (const float*)d_in[12];
  const float* qm_b2   = (const float*)d_in[13];
  const float* km_w1   = (const float*)d_in[14];
  const float* km_b1   = (const float*)d_in[15];
  const float* km_w2   = (const float*)d_in[16];
  const float* km_b2   = (const float*)d_in[17];
  const float* vm_w1   = (const float*)d_in[18];
  const float* vm_b1   = (const float*)d_in[19];
  const float* vm_w2   = (const float*)d_in[20];
  const float* vm_b2   = (const float*)d_in[21];
  const float* ttab    = (const float*)d_in[22];
  const float* tl1_w   = (const float*)d_in[23];
  const float* tl1_b   = (const float*)d_in[24];
  const float* tl2_w   = (const float*)d_in[25];
  const float* tl2_b   = (const float*)d_in[26];
  const float* pw_w1   = (const float*)d_in[27];
  const float* pw_b1   = (const float*)d_in[28];
  const float* pw_w2   = (const float*)d_in[29];
  const float* pw_b2   = (const float*)d_in[30];
  const float* out_lin = (const float*)d_in[31];
  // layers at 32 + 7*lp: w_q, w_k, w_v, proj.w, proj.b, lin.w, lin.b

  // workspace layout (fp32)
  float* ws   = (float*)d_ws;
  float* a_x  = ws;                      // 8,388,608
  float* h    = a_x + 8388608;           // 8,388,608 (MLP hidden scratch)
  float* q_x  = h + 8388608;             // 8,388,608
  float* k_x  = q_x + 8388608;           // 8,388,608
  float* v_x  = k_x + 8388608;           // 8,388,608
  float* aout = v_x + 8388608;           // 4,194,304 attention output
  float* e_t  = aout + 4194304;          // 524,288
  float* mbuf = e_t + 524288;            // 4,096
  float* logs = mbuf + 4096;             // 32,768
  // aliases over dead buffers
  float* qsb  = a_x;                     // qs/ks overlay a_x after mappings
  float* ksb  = a_x + 4194304;
  float* vsb  = h;                       // vs overlays hidden scratch
  float* pout = h + 4194304;             // proj output
  float* traw = h;                       // time pipeline runs before mappings
  float* thid = h + 524288;

  auto GEMM = [&](const float* A, const float* W, const float* bias,
                  const float* Cin, float* C, int Mm, int Nn, int Kk, int relu) {
    gemm16x64_bf16<<<dim3((unsigned)((Mm / 16) * (Nn / 64))), dim3(32), 0, stream>>>(
        A, W, bias, Cin, C, Mm, Nn, Kk, relu);
  };

  const long totElem = (long)Mr * 256;
  const unsigned ebl = (unsigned)((totElem + 255) / 256);

  // 1. embeddings / master / time pipeline
  embed_k<<<dim3(Mr), dim3(256), 0, stream>>>(x, v_emb, a_x);
  master_k<<<dim3(16), dim3(256), 0, stream>>>(master, me_w, me_b, mbuf);
  timegather_k<<<dim3(16 * 128), dim3(256), 0, stream>>>(timep, ttab, traw);
  GEMM(traw, tl1_w, tl1_b, nullptr, thid, 2048, 256, 256, 1);
  GEMM(thid, tl2_w, tl2_b, nullptr, e_t, 2048, 256, 256, 0);

  // 2. gate: a_x += sigmoid(k*m)*m, k = attention_mapping MLP (k staged in q_x)
  GEMM(a_x, am_w1, am_b1, nullptr, h, Mr, 256, 256, 1);
  GEMM(h, am_w2, am_b2, nullptr, q_x, Mr, 256, 256, 0);
  gate_k<<<dim3(ebl), dim3(256), 0, stream>>>(a_x, q_x, mbuf, totElem);

  // 3. q/k/v mappings
  GEMM(a_x, qm_w1, qm_b1, nullptr, h, Mr, 256, 256, 1);
  GEMM(h, qm_w2, qm_b2, nullptr, q_x, Mr, 256, 256, 0);
  GEMM(a_x, km_w1, km_b1, nullptr, h, Mr, 256, 256, 1);
  GEMM(h, km_w2, km_b2, nullptr, k_x, Mr, 256, 256, 0);
  GEMM(a_x, vm_w1, vm_b1, nullptr, h, Mr, 256, 256, 1);
  GEMM(h, vm_w2, vm_b2, nullptr, v_x, Mr, 256, 256, 0);
  addet_k<<<dim3(ebl), dim3(256), 0, stream>>>(q_x, e_t, totElem);

  // 4. attention layers
  for (int lp = 0; lp < 2; lp++) {
    const float* w_q  = (const float*)d_in[32 + 7 * lp + 0];
    const float* w_k  = (const float*)d_in[32 + 7 * lp + 1];
    const float* w_v  = (const float*)d_in[32 + 7 * lp + 2];
    const float* pr_w = (const float*)d_in[32 + 7 * lp + 3];
    const float* pr_b = (const float*)d_in[32 + 7 * lp + 4];
    const float* ln_w = (const float*)d_in[32 + 7 * lp + 5];
    const float* ln_b = (const float*)d_in[32 + 7 * lp + 6];

    addet_k<<<dim3(ebl), dim3(256), 0, stream>>>(k_x, e_t, totElem);
    GEMM(q_x, w_q, nullptr, nullptr, qsb, Mr, 128, 256, 0);
    GEMM(k_x, w_k, nullptr, nullptr, ksb, Mr, 128, 256, 0);
    GEMM(v_x, w_v, nullptr, nullptr, vsb, Mr, 128, 256, 0);
    attn_argmax<<<dim3(L / 16, 16), dim3(32), 0, stream>>>(qsb, ksb, vsb, aout, L);
    GEMM(aout, pr_w, pr_b, nullptr, pout, Mr, 256, 128, 0);
    // concat(q_x, proj) @ lin.w == q_x @ top_half + proj @ bottom_half
    GEMM(q_x, ln_w, ln_b, nullptr, k_x, Mr, 256, 256, 0);
    GEMM(pout, ln_w + 256 * 256, nullptr, k_x, k_x, Mr, 256, 256, 0);
  }

  // 5. attention pooling + output
  GEMM(k_x, pw_w1, pw_b1, nullptr, h, Mr, 256, 256, 1);
  pawl2_k<<<dim3(Mr / 256), dim3(256), 0, stream>>>(h, pw_w2, pw_b2, logs, Mr);
  pool_final_k<<<dim3(16), dim3(256), 0, stream>>>(logs, k_x, out_lin,
                                                   (float*)d_out, L);
}
